// MultiBlocks_53687091200659
// MI455X (gfx1250) — matrix operand order repaired and checked by reading, not compiled, not run on hardware
//
#include <hip/hip_runtime.h>
#include <math.h>

typedef __bf16 bf16;
typedef __attribute__((ext_vector_type(8)))  __bf16 v8bf_t;
typedef __attribute__((ext_vector_type(16))) __bf16 v16bf_t;
typedef __attribute__((ext_vector_type(8)))  float  v8f_t;

#define GF_RELU  1
#define GF_ACC   2
#define GF_BIAS  4
#define GF_BNMAJ 8

// ---------------------------------------------------------------------------
// Batched GEMM: C[M,N] = act( scale * A[M,K] @ op(B) + (bias) (+ Cprev) )
//   KMAJOR (default): B operand is W[N,K], K contiguous  (x @ W^T form)
//   BNMAJ:            B operand is B[K,N], N contiguous  (att @ V form)
// Wave32 WMMA bf16 16x16x32, f32 accumulate. Each wave: 32(M) x 64(N) tile
// (8 accumulators, 2 A fragments, 4 shared B fragments -> 2 WMMAs per B load).
// Block (8 waves) covers 128 x 128. All dims must divide tiles (EXEC all-ones).
// Batch decomposed as z = zb*nH + zh with independent strides (for B,H heads).
// ---------------------------------------------------------------------------
template<int FLAGS>
__global__ __launch_bounds__(256)
void gemm_bf16_wmma(const bf16* __restrict__ A, long ldA, long sAb, long sAh,
                    const bf16* __restrict__ Bm, long ldB, long sBb, long sBh,
                    const float* __restrict__ bias, float scale,
                    float* __restrict__ Cf, bf16* __restrict__ Cb,
                    long ldC, long sCb, long sCh,
                    int M, int N, int K, int nH)
{
  const int z  = blockIdx.z;
  const int zb = z / nH, zh = z - zb * nH;
  A  += (long)zb * sAb + (long)zh * sAh;
  Bm += (long)zb * sBb + (long)zh * sBh;
  const long cOff = (long)zb * sCb + (long)zh * sCh;

  const int lane = threadIdx.x & 31;
  const int wave = threadIdx.x >> 5;
  const int wm   = wave & 3;          // 4 waves along M
  const int wn   = wave >> 2;         // 2 waves along N
  const int m0   = blockIdx.y * 128 + wm * 32;
  const int n0   = blockIdx.x * 128 + wn * 64;
  const int l15  = lane & 15;
  const int lhi  = lane >> 4;

  v8f_t acc[2][4] = {};

  // A fragment row pointers (contiguous k order, same as B):
  // lanes 0-15 take K[k0..k0+15], lanes 16-31 K[k0+16..31].
  const bf16* aRow0 = A + (long)(m0 + l15) * ldA + (lhi << 4);
  const bf16* aRow1 = aRow0 + 16 * ldA;

  for (int k0 = 0; k0 < K; k0 += 32) {
    // Prefetch next K tile (speculative; dropped if invalid).
    __builtin_prefetch(aRow0 + k0 + 64, 0, 1);
    __builtin_prefetch(aRow1 + k0 + 64, 0, 1);

    v16bf_t a0, a1;
    {
      v8bf_t lo = *(const v8bf_t*)(aRow0 + k0);
      v8bf_t hi = *(const v8bf_t*)(aRow0 + k0 + 8);
      #pragma unroll
      for (int i = 0; i < 8; ++i) { a0[i] = lo[i]; a0[8 + i] = hi[i]; }
    }
    {
      v8bf_t lo = *(const v8bf_t*)(aRow1 + k0);
      v8bf_t hi = *(const v8bf_t*)(aRow1 + k0 + 8);
      #pragma unroll
      for (int i = 0; i < 8; ++i) { a1[i] = lo[i]; a1[8 + i] = hi[i]; }
    }

    // Load all four B fragments up-front so WMMAs overlap outstanding loads.
    v16bf_t bfr[4];
    #pragma unroll
    for (int t = 0; t < 4; ++t) {
      const int n = n0 + t * 16 + l15;
      if constexpr (FLAGS & GF_BNMAJ) {
        // B[K,N]: lane n gathers 16 K-rows (lanes 0-15 K=k0..15, hi K=k0+16..31)
        const bf16* bp = Bm + (long)(k0 + (lhi << 4)) * ldB + n;
        #pragma unroll
        for (int j = 0; j < 16; ++j) bfr[t][j] = bp[(long)j * ldB];
      } else {
        // W[N,K]: lane n reads 16 contiguous K elements of row n
        const bf16* bp = Bm + (long)n * ldB + k0 + (lhi << 4);
        __builtin_prefetch(bp + 64, 0, 1);
        v8bf_t lo = *(const v8bf_t*)bp;
        v8bf_t hi = *(const v8bf_t*)(bp + 8);
        #pragma unroll
        for (int i = 0; i < 8; ++i) { bfr[t][i] = lo[i]; bfr[t][8 + i] = hi[i]; }
      }
    }

    #pragma unroll
    for (int t = 0; t < 4; ++t) {
      acc[0][t] = __builtin_amdgcn_wmma_f32_16x16x32_bf16(
                    false, a0, false, bfr[t], (short)0, acc[0][t], false, false);
      acc[1][t] = __builtin_amdgcn_wmma_f32_16x16x32_bf16(
                    false, a1, false, bfr[t], (short)0, acc[1][t], false, false);
    }
  }

  // Epilogue. C layout: VGPR r -> M = base + r + 8*lhi, N = n0 + t*16 + l15.
  #pragma unroll
  for (int t = 0; t < 4; ++t) {
    const int n = n0 + t * 16 + l15;
    float bv = 0.f;
    if constexpr (FLAGS & GF_BIAS) bv = bias[n];
    #pragma unroll
    for (int u = 0; u < 2; ++u) {
      #pragma unroll
      for (int r = 0; r < 8; ++r) {
        const int m = m0 + u * 16 + r + (lhi << 3);
        const long idx = cOff + (long)m * ldC + n;
        float v = acc[u][t][r] * scale + bv;
        if constexpr (FLAGS & GF_ACC)  v += Cf[idx];
        if constexpr (FLAGS & GF_RELU) v = v > 0.f ? v : 0.f;
        if (Cf) Cf[idx] = v;
        if (Cb) Cb[idx] = (bf16)v;
      }
    }
  }
}

// ---------------------------------------------------------------------------
// Softmax over rows of [rows, n] f32 -> bf16. One wave per row (wave32).
// ---------------------------------------------------------------------------
__global__ __launch_bounds__(256)
void k_softmax_rows(const float* __restrict__ S, bf16* __restrict__ O, int n)
{
  const long row  = (long)blockIdx.x * 8 + (threadIdx.x >> 5);
  const int  lane = threadIdx.x & 31;
  const float* p = S + row * n;
  float mx = -3.402823e38f;
  for (int c = lane; c < n; c += 32) mx = fmaxf(mx, p[c]);
  #pragma unroll
  for (int o = 16; o > 0; o >>= 1) mx = fmaxf(mx, __shfl_xor(mx, o, 32));
  float sum = 0.f;
  for (int c = lane; c < n; c += 32) sum += __expf(p[c] - mx);
  #pragma unroll
  for (int o = 16; o > 0; o >>= 1) sum += __shfl_xor(sum, o, 32);
  const float inv = 1.f / sum;
  bf16* q = O + row * n;
  for (int c = lane; c < n; c += 32) q[c] = (bf16)(__expf(p[c] - mx) * inv);
}

// --------------------------- elementwise helpers ---------------------------
__global__ void k_f2b(const float* __restrict__ x, bf16* __restrict__ y, long n)
{ long i = (long)blockIdx.x * 256 + threadIdx.x; if (i < n) y[i] = (bf16)x[i]; }

__global__ void k_add2_f2b(const float* __restrict__ a, const float* __restrict__ b,
                           bf16* __restrict__ o, long n)
{ long i = (long)blockIdx.x * 256 + threadIdx.x; if (i < n) o[i] = (bf16)(a[i] + b[i]); }

__global__ void k_add2_bf(const bf16* __restrict__ a, const bf16* __restrict__ b,
                          bf16* __restrict__ o, long n)
{ long i = (long)blockIdx.x * 256 + threadIdx.x;
  if (i < n) o[i] = (bf16)((float)a[i] + (float)b[i]); }

__global__ void k_acc3(float* __restrict__ acc, const float* __restrict__ a,
                       const float* __restrict__ b, long n)
{ long i = (long)blockIdx.x * 256 + threadIdx.x; if (i < n) acc[i] += a[i] + b[i]; }

// container update: x = x + a + b ; xb = bf16(x)
__global__ void k_cont3(float* __restrict__ x, const float* __restrict__ a,
                        const float* __restrict__ b, bf16* __restrict__ xb, long n)
{ long i = (long)blockIdx.x * 256 + threadIdx.x;
  if (i < n) { float v = x[i] + a[i] + b[i]; x[i] = v; xb[i] = (bf16)v; } }

// mean over sequence dim: x [B,S,D] bf16 -> mean [B,D] f32 (one thread per (b,d))
__global__ void k_mean(const bf16* __restrict__ x, float* __restrict__ mean,
                       int S_, int D_, long total)
{
  long i = (long)blockIdx.x * 256 + threadIdx.x;
  if (i >= total) return;
  long b = i / D_; int d = (int)(i - b * D_);
  const bf16* p = x + b * (long)S_ * D_ + d;
  float s = 0.f;
  for (int r = 0; r < S_; ++r) s += (float)p[(long)r * D_];
  mean[i] = s * (1.f / (float)S_);
}

// gate = sigmoid(mean @ W^T + b); tiny (8x1024) -> plain VALU
__global__ void k_gate(const float* __restrict__ mean, const float* __restrict__ W,
                       const float* __restrict__ bias, float* __restrict__ g,
                       int Din, int Dout, long total)
{
  long i = (long)blockIdx.x * 256 + threadIdx.x;
  if (i >= total) return;
  long bb = i / Dout; int n = (int)(i - bb * Dout);
  const float* m = mean + bb * (long)Din;
  const float* w = W + (long)n * Din;
  float s = bias[n];
  for (int k = 0; k < Din; ++k) s += m[k] * w[k];
  g[i] = 1.f / (1.f + __expf(-s));
}

// x[b,s,c] *= (1 + gate[b, c % D]) over [B,S,C] bf16
__global__ void k_applygate(bf16* __restrict__ x, const float* __restrict__ gate,
                            int S_, int C_, int D_, long total)
{
  long i = (long)blockIdx.x * 256 + threadIdx.x;
  if (i >= total) return;
  long b = i / ((long)S_ * C_);
  int  c = (int)(i % C_);
  float g = 1.f + gate[b * (long)D_ + (c % D_)];
  x[i] = (bf16)((float)x[i] * g);
}

// ------------------------------ host helpers -------------------------------
static void run_gemm(hipStream_t st, int flags,
                     const bf16* A, long ldA, long sAb, long sAh,
                     const bf16* Bm, long ldB, long sBb, long sBh,
                     const float* bias, float scale,
                     float* Cf, bf16* Cb, long ldC, long sCb, long sCh,
                     int M, int N, int K, int nH, int nBatch)
{
  dim3 grid((unsigned)(N / 128), (unsigned)(M / 128), (unsigned)nBatch);
  dim3 blk(256);
#define LG(F) gemm_bf16_wmma<F><<<grid, blk, 0, st>>>(A, ldA, sAb, sAh, Bm, ldB, sBb, sBh, \
              bias, scale, Cf, Cb, ldC, sCb, sCh, M, N, K, nH)
  if      (flags == 0)                              LG(0);
  else if (flags == (GF_RELU | GF_BIAS))            LG(GF_RELU | GF_BIAS);
  else if (flags == (GF_ACC | GF_BIAS | GF_RELU))   LG(GF_ACC | GF_BIAS | GF_RELU);
  else                                              LG(GF_BNMAJ);
#undef LG
}

extern "C" void kernel_launch(void* const* d_in, const int* in_sizes, int n_in,
                              void* d_out, int out_size, void* d_ws, size_t ws_size,
                              hipStream_t stream)
{
  (void)in_sizes; (void)n_in; (void)out_size; (void)ws_size;
  const int Bt = 8, S = 512, Din = 512, D = 1024, T3 = 3072, H = 8, DH = 128;
  const int Mr = Bt * S;                 // 4096
  const int BH = Bt * H;                 // 64
  const long ND  = (long)Mr * D;         // 4M elements
  const long ND3 = (long)Mr * T3;
  const long NSC = (long)BH * S * S;     // 16M elements

  const float* v_in = (const float*)d_in[0];
  const float* q_in = (const float*)d_in[1];
  auto Wf = [&](int i) { return (const float*)d_in[i]; };

  char* base = (char*)d_ws;
  size_t off = 0;
  auto alloc = [&](size_t bytes) -> void* {
    off = (off + 255) & ~(size_t)255;
    void* p = base + off; off += bytes; return p;
  };

  // ---- bf16 weight cache (GEMM weights only; gate weights stay f32) ----
  struct WSlot { int idx, N, K; bf16* p; };
  WSlot ws_[18]; int nw = 0;
  ws_[nw++] = {2, D, Din, nullptr};            // init v_lin
  ws_[nw++] = {4, D, Din, nullptr};            // init q_lin
  for (int blk = 0; blk < 2; ++blk) {
    int ib = 6 + blk * 20;
    ws_[nw++] = {ib + 0,  3 * D, D,     nullptr};  // inter v_lin
    ws_[nw++] = {ib + 2,  3 * D, D,     nullptr};  // inter q_lin
    ws_[nw++] = {ib + 4,  D,     2 * D, nullptr};  // inter v_out
    ws_[nw++] = {ib + 6,  D,     2 * D, nullptr};  // inter q_out
    ws_[nw++] = {ib + 12, 3 * D, D,     nullptr};  // intra v_lin
    ws_[nw++] = {ib + 14, 3 * D, D,     nullptr};  // intra q_lin
    ws_[nw++] = {ib + 16, D,     D,     nullptr};  // intra v_out
    ws_[nw++] = {ib + 18, D,     D,     nullptr};  // intra q_out
  }
  for (int i = 0; i < nw; ++i) {
    long n = (long)ws_[i].N * ws_[i].K;
    ws_[i].p = (bf16*)alloc((size_t)n * sizeof(bf16));
    k_f2b<<<(unsigned)((n + 255) / 256), 256, 0, stream>>>(Wf(ws_[i].idx), ws_[i].p, n);
  }
  auto WB = [&](int s) { return ws_[s].p; };

  // ---- activations / scratch ----
  float* v0f  = (float*)alloc(ND * 4);
  float* q0f  = (float*)alloc(ND * 4);
  float* v1f  = (float*)alloc(ND * 4);
  float* q1f  = (float*)alloc(ND * 4);
  float* v2f  = (float*)alloc(ND * 4);
  float* q2f  = (float*)alloc(ND * 4);
  float* accv = (float*)alloc(ND * 4);
  float* accq = (float*)alloc(ND * 4);
  float* scoresf = (float*)alloc((size_t)NSC * 4);
  bf16* v0b = (bf16*)alloc(ND * 2);
  bf16* q0b = (bf16*)alloc(ND * 2);
  bf16* vtb = (bf16*)alloc((size_t)ND3 * 2);
  bf16* qtb = (bf16*)alloc((size_t)ND3 * 2);
  bf16* attb = (bf16*)alloc((size_t)NSC * 2);
  bf16* vub = (bf16*)alloc(ND * 2);
  bf16* qub = (bf16*)alloc(ND * 2);
  bf16* vIb = (bf16*)alloc(ND * 2);
  bf16* qIb = (bf16*)alloc(ND * 2);
  bf16* vsb = (bf16*)alloc(ND * 2);
  bf16* qsb = (bf16*)alloc(ND * 2);
  bf16* vinb = (bf16*)alloc((size_t)Mr * Din * 2);
  bf16* qinb = (bf16*)alloc((size_t)Mr * Din * 2);
  float* vmean = (float*)alloc((size_t)Bt * D * 4);
  float* qmean = (float*)alloc((size_t)Bt * D * 4);
  float* gateV = (float*)alloc((size_t)Bt * D * 4);  // applied to v path (from q_mean, q4v)
  float* gateQ = (float*)alloc((size_t)Bt * D * 4);  // applied to q path (from v_mean, v4q)

  const float sc = 1.f / sqrtf((float)DH);
  const long sBH = (long)H * S * S, sSS = (long)S * S, sST3 = (long)S * T3, sSD = (long)S * D;
  const unsigned gE = (unsigned)((ND + 255) / 256);

  // ---- initial projections: v0 = relu(v @ W0^T + b0), q0 likewise ----
  k_f2b<<<(unsigned)(((long)Mr * Din + 255) / 256), 256, 0, stream>>>(v_in, vinb, (long)Mr * Din);
  k_f2b<<<(unsigned)(((long)Mr * Din + 255) / 256), 256, 0, stream>>>(q_in, qinb, (long)Mr * Din);
  run_gemm(stream, GF_RELU | GF_BIAS, vinb, Din, 0, 0, WB(0), Din, 0, 0, Wf(3), 1.f,
           v0f, v0b, D, 0, 0, Mr, D, Din, 1, 1);
  run_gemm(stream, GF_RELU | GF_BIAS, qinb, Din, 0, 0, WB(1), Din, 0, 0, Wf(5), 1.f,
           q0f, q0b, D, 0, 0, Mr, D, Din, 1, 1);
  hipMemcpyAsync(accv, v0f, (size_t)ND * 4, hipMemcpyDeviceToDevice, stream);
  hipMemcpyAsync(accq, q0f, (size_t)ND * 4, hipMemcpyDeviceToDevice, stream);

  for (int blk = 0; blk < 2; ++blk) {
    const int ib = 6 + blk * 20;   // d_in param base
    const int bs = 2 + blk * 8;    // bf16 weight slot base

    // ================= inter =================
    run_gemm(stream, GF_RELU | GF_BIAS, v0b, D, 0, 0, WB(bs + 0), D, 0, 0, Wf(ib + 1), 1.f,
             nullptr, vtb, T3, 0, 0, Mr, 3 * D, D, 1, 1);
    run_gemm(stream, GF_RELU | GF_BIAS, q0b, D, 0, 0, WB(bs + 1), D, 0, 0, Wf(ib + 3), 1.f,
             nullptr, qtb, T3, 0, 0, Mr, 3 * D, D, 1, 1);
    // v_update = mha(v_query, q_key, q_val)
    run_gemm(stream, 0, vtb + D, T3, sST3, DH, qtb, T3, sST3, DH, nullptr, sc,
             scoresf, nullptr, S, sBH, sSS, S, S, DH, H, BH);
    k_softmax_rows<<<(unsigned)(BH * S / 8), 256, 0, stream>>>(scoresf, attb, S);
    run_gemm(stream, GF_BNMAJ, attb, S, sBH, sSS, qtb + 2 * D, T3, sST3, DH, nullptr, 1.f,
             nullptr, vub, D, sSD, DH, S, DH, S, H, BH);
    // q_update = mha(q_query, v_key, v_val)
    run_gemm(stream, 0, qtb + D, T3, sST3, DH, vtb, T3, sST3, DH, nullptr, sc,
             scoresf, nullptr, S, sBH, sSS, S, S, DH, H, BH);
    k_softmax_rows<<<(unsigned)(BH * S / 8), 256, 0, stream>>>(scoresf, attb, S);
    run_gemm(stream, GF_BNMAJ, attb, S, sBH, sSS, vtb + 2 * D, T3, sST3, DH, nullptr, 1.f,
             nullptr, qub, D, sSD, DH, S, DH, S, H, BH);
    // v1 = relu([v0 | v_update] @ Wvout^T + b) as two K=1024 passes
    run_gemm(stream, 0, v0b, D, 0, 0, WB(bs + 2), 2 * D, 0, 0, nullptr, 1.f,
             v1f, nullptr, D, 0, 0, Mr, D, D, 1, 1);
    run_gemm(stream, GF_ACC | GF_BIAS | GF_RELU, vub, D, 0, 0, WB(bs + 2) + D, 2 * D, 0, 0,
             Wf(ib + 5), 1.f, v1f, nullptr, D, 0, 0, Mr, D, D, 1, 1);
    run_gemm(stream, 0, q0b, D, 0, 0, WB(bs + 3), 2 * D, 0, 0, nullptr, 1.f,
             q1f, nullptr, D, 0, 0, Mr, D, D, 1, 1);
    run_gemm(stream, GF_ACC | GF_BIAS | GF_RELU, qub, D, 0, 0, WB(bs + 3) + D, 2 * D, 0, 0,
             Wf(ib + 7), 1.f, q1f, nullptr, D, 0, 0, Mr, D, D, 1, 1);

    // ================= intra (input vI = v1 + v0, qI = q1 + q0) =================
    k_add2_f2b<<<gE, 256, 0, stream>>>(v0f, v1f, vIb, ND);
    k_add2_f2b<<<gE, 256, 0, stream>>>(q0f, q1f, qIb, ND);
    k_mean<<<(unsigned)((Bt * D + 255) / 256), 256, 0, stream>>>(vIb, vmean, S, D, (long)Bt * D);
    k_mean<<<(unsigned)((Bt * D + 255) / 256), 256, 0, stream>>>(qIb, qmean, S, D, (long)Bt * D);
    // gv = 1 + sigmoid(fc(q_mean, q4v)) applied to v;  gq = 1 + sigmoid(fc(v_mean, v4q)) applied to q
    k_gate<<<(unsigned)((Bt * D + 255) / 256), 256, 0, stream>>>(qmean, Wf(ib + 10), Wf(ib + 11), gateV, D, D, (long)Bt * D);
    k_gate<<<(unsigned)((Bt * D + 255) / 256), 256, 0, stream>>>(vmean, Wf(ib + 8),  Wf(ib + 9),  gateQ, D, D, (long)Bt * D);
    run_gemm(stream, GF_RELU | GF_BIAS, vIb, D, 0, 0, WB(bs + 4), D, 0, 0, Wf(ib + 13), 1.f,
             nullptr, vtb, T3, 0, 0, Mr, 3 * D, D, 1, 1);
    run_gemm(stream, GF_RELU | GF_BIAS, qIb, D, 0, 0, WB(bs + 5), D, 0, 0, Wf(ib + 15), 1.f,
             nullptr, qtb, T3, 0, 0, Mr, 3 * D, D, 1, 1);
    k_applygate<<<(unsigned)((ND3 + 255) / 256), 256, 0, stream>>>(vtb, gateV, S, T3, D, ND3);
    k_applygate<<<(unsigned)((ND3 + 255) / 256), 256, 0, stream>>>(qtb, gateQ, S, T3, D, ND3);
    // v self-attention
    run_gemm(stream, 0, vtb + D, T3, sST3, DH, vtb, T3, sST3, DH, nullptr, sc,
             scoresf, nullptr, S, sBH, sSS, S, S, DH, H, BH);
    k_softmax_rows<<<(unsigned)(BH * S / 8), 256, 0, stream>>>(scoresf, attb, S);
    run_gemm(stream, GF_BNMAJ, attb, S, sBH, sSS, vtb + 2 * D, T3, sST3, DH, nullptr, 1.f,
             nullptr, vub, D, sSD, DH, S, DH, S, H, BH);
    // q self-attention
    run_gemm(stream, 0, qtb + D, T3, sST3, DH, qtb, T3, sST3, DH, nullptr, sc,
             scoresf, nullptr, S, sBH, sSS, S, S, DH, H, BH);
    k_softmax_rows<<<(unsigned)(BH * S / 8), 256, 0, stream>>>(scoresf, attb, S);
    run_gemm(stream, GF_BNMAJ, attb, S, sBH, sSS, qtb + 2 * D, T3, sST3, DH, nullptr, 1.f,
             nullptr, qub, D, sSD, DH, S, DH, S, H, BH);
    // v2 = relu((vI + v_update) @ W^T + b)
    k_add2_bf<<<gE, 256, 0, stream>>>(vIb, vub, vsb, ND);
    k_add2_bf<<<gE, 256, 0, stream>>>(qIb, qub, qsb, ND);
    run_gemm(stream, GF_RELU | GF_BIAS, vsb, D, 0, 0, WB(bs + 6), D, 0, 0, Wf(ib + 17), 1.f,
             v2f, nullptr, D, 0, 0, Mr, D, D, 1, 1);
    run_gemm(stream, GF_RELU | GF_BIAS, qsb, D, 0, 0, WB(bs + 7), D, 0, 0, Wf(ib + 19), 1.f,
             q2f, nullptr, D, 0, 0, Mr, D, D, 1, 1);

    // results: acc += v1 + v2 ; container: v0 = v0 + v1 + v2 (and refresh bf16)
    k_acc3<<<gE, 256, 0, stream>>>(accv, v1f, v2f, ND);
    k_acc3<<<gE, 256, 0, stream>>>(accq, q1f, q2f, ND);
    k_cont3<<<gE, 256, 0, stream>>>(v0f, v1f, v2f, v0b, ND);
    k_cont3<<<gE, 256, 0, stream>>>(q0f, q1f, q2f, q0b, ND);
  }

  float* outp = (float*)d_out;
  hipMemcpyAsync(outp,      accv, (size_t)ND * 4, hipMemcpyDeviceToDevice, stream);
  hipMemcpyAsync(outp + ND, accq, (size_t)ND * 4, hipMemcpyDeviceToDevice, stream);
}
